// LinearAttentionParallel_16990890623046
// MI455X (gfx1250) — compile-verified, matrix operand order checked by reading, not run on hardware
//
#include <hip/hip_runtime.h>
#include <hip/hip_bf16.h>

// ---------------------------------------------------------------------------
// LinearAttentionParallel on MI455X (gfx1250, wave32, WMMA bf16)
//   B=8, T=2048, D=1024, E=3D=3072
// ---------------------------------------------------------------------------

typedef __bf16 bf16_t;
typedef __attribute__((ext_vector_type(16))) __bf16 v16bf;
typedef __attribute__((ext_vector_type(8)))  __bf16 v8bf;
typedef __attribute__((ext_vector_type(8)))  float  v8f;

constexpr int NB = 8;
constexpr int TT = 2048;
constexpr int DD = 1024;
constexpr int EE = 3 * DD;

// ---- optional gfx1250 async-to-LDS path (guarded; falls back cleanly) ------
#if defined(__has_builtin)
#  if __has_builtin(__builtin_amdgcn_global_load_async_to_lds_b128)
#    define USE_ASYNC_LDS 1
#  endif
#  if __has_builtin(__builtin_amdgcn_s_wait_asynccnt)
#    define HAVE_WAIT_ASYNC_BUILTIN 1
#  endif
#endif
#ifndef USE_ASYNC_LDS
#  define USE_ASYNC_LDS 0
#endif

#if USE_ASYNC_LDS
// Exact parameter types per hipcc diagnostic: int4 (gcc vector_size) pointers,
// global (AS1) source and LDS (AS3) destination.
typedef int v4i __attribute__((vector_size(16)));
typedef __attribute__((address_space(1))) v4i* gv4i_p;
typedef __attribute__((address_space(3))) v4i* lv4i_p;
#endif

static __device__ __forceinline__ void wait_asynccnt0() {
#if defined(HAVE_WAIT_ASYNC_BUILTIN)
  __builtin_amdgcn_s_wait_asynccnt(0);
#else
  asm volatile("s_wait_asynccnt 0x0" ::: "memory");
#endif
}

// ---- WMMA helper -----------------------------------------------------------
static __device__ __forceinline__ v8f wmma_bf16(v16bf a, v16bf b, v8f c) {
  // 8 args: (neg_a, A, neg_b, B, c_mod, C, reuse_a, reuse_b)
  return __builtin_amdgcn_wmma_f32_16x16x32_bf16(false, a, false, b,
                                                 (short)0, c, false, false);
}

// A/B fragment from bf16 memory: halves at p[off..off+7] and p[off+16..off+23]
// (ISA 16-bit 16x32 layout: lane holds K {kb..kb+7, 16+kb..16+kb+7}, kb=8*(lane>=16))
static __device__ __forceinline__ v16bf load_frag_bf16(const bf16_t* __restrict__ p,
                                                       int off) {
  v8bf lo = *(const v8bf*)(p + off);
  v8bf hi = *(const v8bf*)(p + off + 16);
  v16bf r;
#pragma unroll
  for (int t = 0; t < 8; ++t) { r[t] = lo[t]; r[8 + t] = hi[t]; }
  return r;
}

// ---------------------------------------------------------------------------
// Kernel 0: one-time f32 -> bf16 downconvert (8 elems / thread).
// ---------------------------------------------------------------------------
__global__ __launch_bounds__(256)
void cast_bf16_kernel(const float* __restrict__ src, bf16_t* __restrict__ dst) {
  const size_t i = ((size_t)blockIdx.x * 256 + threadIdx.x) * 8;
  float4 a = *(const float4*)(src + i);
  float4 b = *(const float4*)(src + i + 4);
  v8bf o;
  o[0] = (bf16_t)a.x; o[1] = (bf16_t)a.y; o[2] = (bf16_t)a.z; o[3] = (bf16_t)a.w;
  o[4] = (bf16_t)b.x; o[5] = (bf16_t)b.y; o[6] = (bf16_t)b.z; o[7] = (bf16_t)b.w;
  *(v8bf*)(dst + i) = o;
}

// ---------------------------------------------------------------------------
// Kernel 1: qkv[M=B*T, E] = bf16( xb[M,D] @ Wb[E,D]^T + bias )
// One wave computes a 16x64 strip: A fragment reused over 4 N-tiles.
// Inner loop is pure  global_load_b128 -> v_wmma  (bf16 operands pre-converted).
// ---------------------------------------------------------------------------
__global__ __launch_bounds__(32)
void qkv_gemm_kernel(const bf16_t* __restrict__ xb, const bf16_t* __restrict__ Wb,
                     const float* __restrict__ bqkv, bf16_t* __restrict__ qkv) {
  const int lane = threadIdx.x;
  const int row0 = blockIdx.x * 16;       // M tile
  const int n0   = blockIdx.y * 64;       // N strip (4 tiles of 16)
  const int rsel = lane & 15;
  const int kb   = (lane >> 4) * 8;

  const bf16_t* __restrict__ Arow = xb + (size_t)(row0 + rsel) * DD;
  const bf16_t* __restrict__ Brow[4];
#pragma unroll
  for (int nt = 0; nt < 4; ++nt)
    Brow[nt] = Wb + (size_t)(n0 + nt * 16 + rsel) * DD;

  v8f acc[4];
#pragma unroll
  for (int nt = 0; nt < 4; ++nt)
#pragma unroll
    for (int r = 0; r < 8; ++r) acc[nt][r] = 0.0f;

  for (int k = 0; k < DD; k += 32) {
    if (k + 64 < DD) __builtin_prefetch(Arow + k + 64 + kb, 0, 1);
    v16bf a = load_frag_bf16(Arow, k + kb);
#pragma unroll
    for (int nt = 0; nt < 4; ++nt) {
      v16bf b = load_frag_bf16(Brow[nt], k + kb);
      acc[nt] = wmma_bf16(a, b, acc[nt]);
    }
  }

  // C layout: lanes 0-15 -> M=r, lanes 16-31 -> M=8+r ; N = lane&15
#pragma unroll
  for (int nt = 0; nt < 4; ++nt) {
    const int n = n0 + nt * 16 + rsel;
    const float bias = bqkv[n];
#pragma unroll
    for (int r = 0; r < 8; ++r) {
      const int m = r + ((lane >> 4) << 3);
      qkv[(size_t)(row0 + m) * EE + n] = (bf16_t)(acc[nt][r] + bias);
    }
  }
}

// ---------------------------------------------------------------------------
// Kernel 2: Vt[b][d][t] = qkv[b*T+t][2D + d]  (LDS-tiled 32x32 transpose)
// ---------------------------------------------------------------------------
__global__ __launch_bounds__(256)
void vtrans_kernel(const bf16_t* __restrict__ qkv, bf16_t* __restrict__ vt) {
  __shared__ bf16_t tile[32][33];
  const int b  = blockIdx.z;
  const int t0 = blockIdx.x * 32;
  const int d0 = blockIdx.y * 32;
  const int tx = threadIdx.x;   // 0..31
  const int ty = threadIdx.y;   // 0..7
#pragma unroll
  for (int s = 0; s < 4; ++s) {
    const int t = ty + s * 8;
    tile[t][tx] = qkv[(size_t)(b * TT + t0 + t) * EE + 2 * DD + d0 + tx];
  }
  __syncthreads();
#pragma unroll
  for (int s = 0; s < 4; ++s) {
    const int d = ty + s * 8;
    vt[(size_t)(b * DD + d0 + d) * TT + t0 + tx] = tile[tx][d];
  }
}

// ---------------------------------------------------------------------------
// Kernel 3: causal linear attention, flash-style.
// Workgroup = 16 waves; owns a 16-row i-block. Wave w owns D-slice [64w,64w+64).
// Per 32-wide causal j-step:
//   (async: launch V-slice copy global->LDS, overlapped with S compute)
//   S_partial (per-wave QK^T over D-slice) -> LDS reduce -> mask/scale ->
//   bf16 S in LDS -> each wave: Y[16x64] += S @ V.
// ---------------------------------------------------------------------------
__global__ __launch_bounds__(512)
void attn_kernel(const bf16_t* __restrict__ qkv, const bf16_t* __restrict__ vt,
                 float* __restrict__ y) {
  __shared__ float Spart[16 * 512];                 // 32 KB partial S (16 waves)
  __shared__ __align__(16) bf16_t Sb[512];          // reduced S tile (16x32 bf16)
#if USE_ASYNC_LDS
  __shared__ __align__(16) bf16_t Vbuf[16 * 64 * 32]; // 64 KB: per-wave 64d x 32j
#endif

  const int tid  = threadIdx.x;
  const int w    = tid >> 5;
  const int lane = tid & 31;
  const int rsel = lane & 15;
  const int kb   = (lane >> 4) * 8;
  const int b    = blockIdx.y;
  const int i0   = blockIdx.x * 16;
  const int dw   = w * 64;

  const float inv_scale = 6.9053396600248781e-04f;  // 1/sqrt(D*T) = 2^-10.5

  // ---- hoisted Q fragments: loop-invariant across the causal j-loop ----
  const bf16_t* __restrict__ Qrow =
      qkv + (size_t)(b * TT + i0 + rsel) * EE + dw;
  v16bf qf[2];
#pragma unroll
  for (int h = 0; h < 2; ++h) qf[h] = load_frag_bf16(Qrow + h * 32, kb);

  v8f yacc[4];
#pragma unroll
  for (int nt = 0; nt < 4; ++nt)
#pragma unroll
    for (int r = 0; r < 8; ++r) yacc[nt][r] = 0.0f;

  const int jsteps = i0 / 32 + 1;                   // causal: only j0 <= i0
  for (int js = 0; js < jsteps; ++js) {
    const int j0 = js * 32;

#if USE_ASYNC_LDS
    // ---- phase 0: async-stage this wave's V slice (64 rows x 32 j, 4 KB)
    //      into LDS; overlaps with the QK^T WMMAs below (ASYNCcnt-tracked).
#pragma unroll
    for (int it = 0; it < 8; ++it) {
      const int c   = it * 32 + lane;               // 256 16-byte chunks
      const int dl  = c >> 2;                       // local d row (0..63)
      const int jo  = (c & 3) * 8;                  // j offset within row
      const size_t goff = (size_t)(b * DD + dw + dl) * TT + j0 + jo;
      const int    loff = w * 2048 + dl * 32 + jo;
      __builtin_amdgcn_global_load_async_to_lds_b128(
          (gv4i_p)(bf16_t*)(vt + goff), (lv4i_p)&Vbuf[loff], 0, 0);
    }
#endif

    // ---- phase 1: per-wave partial S (16 x 32) over its 64-wide D slice ----
    v8f s0, s1;
#pragma unroll
    for (int r = 0; r < 8; ++r) { s0[r] = 0.0f; s1[r] = 0.0f; }
#pragma unroll
    for (int dd = 0; dd < 64; dd += 32) {
      const v16bf a = qf[dd >> 5];
      {
        const bf16_t* Krow =
            qkv + (size_t)(b * TT + j0 + rsel) * EE + DD + dw + dd;
        s0 = wmma_bf16(a, load_frag_bf16(Krow, kb), s0);
      }
      {
        const bf16_t* Krow =
            qkv + (size_t)(b * TT + j0 + 16 + rsel) * EE + DD + dw + dd;
        s1 = wmma_bf16(a, load_frag_bf16(Krow, kb), s1);
      }
    }

    // ---- phase 2: spill partial S to LDS ----
    float* sp = &Spart[w * 512];
#pragma unroll
    for (int r = 0; r < 8; ++r) {
      const int m = r + ((lane >> 4) << 3);
      sp[m * 32 + rsel]      = s0[r];
      sp[m * 32 + 16 + rsel] = s1[r];
    }
    __syncthreads();

    // ---- phase 3: reduce 16 partials, mask+scale, requantize to bf16 ----
    {
      float s = 0.0f;
#pragma unroll
      for (int ww = 0; ww < 16; ++ww) s += Spart[ww * 512 + tid];
      const int m = tid >> 5, n = tid & 31;
      const int gi = i0 + m, gj = j0 + n;
      s = (gj <= gi) ? s * inv_scale : 0.0f;
      Sb[tid] = (bf16_t)s;
    }
    __syncthreads();

    // ---- phase 4: Y[16x64] += S(16x32) @ V(32x64) ----
    v16bf af;
#pragma unroll
    for (int t = 0; t < 8; ++t) {
      af[t]     = Sb[rsel * 32 + kb + t];
      af[8 + t] = Sb[rsel * 32 + 16 + kb + t];
    }
#if USE_ASYNC_LDS
    wait_asynccnt0();                               // V slice resident in LDS
#pragma unroll
    for (int nt = 0; nt < 4; ++nt) {
      const int base = w * 2048 + (nt * 16 + rsel) * 32;
      v8bf lo = *(const v8bf*)&Vbuf[base + kb];
      v8bf hi = *(const v8bf*)&Vbuf[base + 16 + kb];
      v16bf bfg;
#pragma unroll
      for (int t = 0; t < 8; ++t) { bfg[t] = lo[t]; bfg[8 + t] = hi[t]; }
      yacc[nt] = wmma_bf16(af, bfg, yacc[nt]);
    }
#else
#pragma unroll
    for (int nt = 0; nt < 4; ++nt) {
      const bf16_t* Vrow = vt + (size_t)(b * DD + dw + nt * 16 + rsel) * TT + j0;
      yacc[nt] = wmma_bf16(af, load_frag_bf16(Vrow, kb), yacc[nt]);
    }
#endif
    __syncthreads();  // protect Spart/Sb for next j-step
  }

  // ---- epilogue: f32 Y to d_out ----
#pragma unroll
  for (int nt = 0; nt < 4; ++nt)
#pragma unroll
    for (int r = 0; r < 8; ++r) {
      const int m = r + ((lane >> 4) << 3);
      y[(size_t)(b * TT + i0 + m) * DD + dw + nt * 16 + rsel] = yacc[nt][r];
    }
}

// ---------------------------------------------------------------------------
// Kernel 4: in-place LayerNorm over last dim (one block per row).
// ---------------------------------------------------------------------------
__global__ __launch_bounds__(256)
void ln_kernel(float* __restrict__ y, const float* __restrict__ lw,
               const float* __restrict__ lb) {
  __shared__ float red1[256], red2[256];
  const int tid = threadIdx.x;
  float* p = y + (size_t)blockIdx.x * DD;

  float v[4], s1 = 0.0f, s2 = 0.0f;
#pragma unroll
  for (int k = 0; k < 4; ++k) {
    v[k] = p[tid + k * 256];
    s1 += v[k];
    s2 += v[k] * v[k];
  }
  red1[tid] = s1; red2[tid] = s2;
  __syncthreads();
  for (int off = 128; off > 0; off >>= 1) {
    if (tid < off) { red1[tid] += red1[tid + off]; red2[tid] += red2[tid + off]; }
    __syncthreads();
  }
  const float mean = red1[0] * (1.0f / DD);
  const float var  = red2[0] * (1.0f / DD) - mean * mean;
  const float rstd = rsqrtf(var + 1e-5f);
#pragma unroll
  for (int k = 0; k < 4; ++k) {
    const int d = tid + k * 256;
    p[d] = (v[k] - mean) * rstd * lw[d] + lb[d];
  }
}

// ---------------------------------------------------------------------------
extern "C" void kernel_launch(void* const* d_in, const int* in_sizes, int n_in,
                              void* d_out, int out_size, void* d_ws, size_t ws_size,
                              hipStream_t stream) {
  (void)in_sizes; (void)n_in; (void)out_size; (void)ws_size;
  const float* x   = (const float*)d_in[0];   // [B,T,D]
  const float* Wq  = (const float*)d_in[1];   // [3D,D]
  const float* bq  = (const float*)d_in[2];   // [3D]
  const float* lnw = (const float*)d_in[3];   // [D]
  const float* lnb = (const float*)d_in[4];   // [D]
  float* out = (float*)d_out;                 // [B,T,D] f32

  char* ws = (char*)d_ws;
  bf16_t* qkv = (bf16_t*)ws;                        ws += (size_t)NB*TT*EE*2; // 96 MiB
  bf16_t* vt  = (bf16_t*)ws;                        ws += (size_t)NB*DD*TT*2; // 32 MiB
  bf16_t* xb  = (bf16_t*)ws;                        ws += (size_t)NB*TT*DD*2; // 32 MiB
  bf16_t* wb  = (bf16_t*)ws;                                                  //  6 MiB

  const int xblk = (NB * TT * DD) / (256 * 8);   // 8192
  const int wblk = (EE * DD) / (256 * 8);        // 1536
  cast_bf16_kernel<<<xblk, 256, 0, stream>>>(x, xb);
  cast_bf16_kernel<<<wblk, 256, 0, stream>>>(Wq, wb);

  qkv_gemm_kernel<<<dim3(NB * TT / 16, EE / 64), 32, 0, stream>>>(xb, wb, bq, qkv);
  vtrans_kernel<<<dim3(TT / 32, DD / 32, NB), dim3(32, 8), 0, stream>>>(qkv, vt);
  attn_kernel<<<dim3(TT / 16, NB), 512, 0, stream>>>(qkv, vt, out);
  ln_kernel<<<NB * TT, 256, 0, stream>>>(out, lnw, lnb);
}
